// RegimeGatedScoreNet_57870389347050
// MI455X (gfx1250) — compile-verified, matrix operand order checked by reading, not run on hardware
//
#include <hip/hip_runtime.h>
#include <hip/hip_bf16.h>

#define DEVI __device__ __forceinline__

typedef __bf16 bf16;
typedef __attribute__((ext_vector_type(8)))  __bf16 bf16x8;
typedef __attribute__((ext_vector_type(16))) __bf16 v16bf;
typedef __attribute__((ext_vector_type(8)))  float  v8f;
typedef __attribute__((ext_vector_type(4)))  unsigned v4u;
typedef __attribute__((ext_vector_type(8)))  unsigned v8u;

#define N_ 512
#define K_ 8
#define H_ 256
#define B_ 8192

#define TDM_CH   128            // K-chunk (elements) DMA'd per step
#define TDM_ROWP 136            // row pitch in elements (128 + 4dw pad = 272B)

// ---------- helpers ----------
DEVI float sigf(float x)  { return 1.f / (1.f + __expf(-x)); }
DEVI float dsilu(float x) { float s = sigf(x); return s * (1.f + x * (1.f - s)); }

// Load one 16x32 bf16 WMMA operand fragment for this lane.
// ISA layout (A and B symmetric in addressing): lane&15 = row(A)/col(B),
// lane>>4 selects K-offset 0 or 8; VGPR0..3 hold K=koff..koff+7, VGPR4..7 hold K=koff+16..+23.
// -> two contiguous 16-byte loads at p and p+16 elements.
DEVI v16bf frag_load(const bf16* p) {
  union { v16bf v; bf16x8 h[2]; } u;
  u.h[0] = *(const bf16x8*)(p);
  u.h[1] = *(const bf16x8*)(p + 16);
  return u.v;
}

// ---------- Tensor Data Mover: DMA a 64-row x 128-elem bf16 panel into LDS ----------
// D# per CDNA5 ISA: group0 = {ctl, lds_addr, global_addr, type},
// group1 = {flags/pads, dims, tile dims, strides}. data_size=1 (2B). pad_enable adds
// 4 dwords after every 64 dwords (one 128-elem row) -> 272B LDS row pitch (bank rotation).
DEVI void tdm_load_tile(const bf16* gsrc, unsigned lds_byte_off, int ldb) {
  unsigned long long ga = (unsigned long long)gsrc;
  v4u g0;
  g0[0] = 1u;                                                  // count=1 (valid user D#)
  g0[1] = lds_byte_off;                                        // lds_addr (bytes)
  g0[2] = (unsigned)ga;                                        // global_addr[31:0]
  g0[3] = (unsigned)((ga >> 32) & 0x01FFFFFFu) | (2u << 30);   // addr[56:32] | type=2
  v8u g1;
  g1[0] = (1u << 16)            // data_size = 1 -> 2 bytes
        | (1u << 20)            // pad_enable
        | (5u << 22)            // pad_interval code 5 -> every 64 dwords
        | (3u << 25);           // pad_amount  code 3 -> 4 dwords (16B)
  g1[1] = 0u;                   // tensor_dim0 = 1<<20: lo16 (<<16) = 0
  g1[2] = 0x10u;                // tensor_dim0 hi16 = 0x10 ; tensor_dim1 lo16 = 0
  g1[3] = 0x10u | (128u << 16); // tensor_dim1 hi16 = 0x10 ; tile_dim0 = 128
  g1[4] = 64u;                  // tile_dim1 = 64 rows ; tile_dim2 = 0
  g1[5] = (unsigned)ldb;        // tensor_dim0_stride lo32 (elements)
  g1[6] = 0u;                   // stride0 hi16 / stride1 lo16
  g1[7] = 0u;                   // stride1 hi32
  asm volatile("tensor_load_to_lds %0, %1" :: "s"(g0), "s"(g1) : "memory");
}

// ---------- prep kernels ----------
__global__ void split_cast_z(const float* __restrict__ z, bf16* __restrict__ x0,
                             bf16* __restrict__ x1) {
  long i = (long)blockIdx.x * blockDim.x + threadIdx.x;
  if (i >= (long)B_ * N_) return;
  long b = i >> 9; int j = (int)(i & (N_ - 1));
  x0[i] = (bf16)z[b * (2 * N_) + j];
  x1[i] = (bf16)z[b * (2 * N_) + N_ + j];
}

__global__ void cast_flat(const float* __restrict__ s, bf16* __restrict__ d, long n) {
  long i = (long)blockIdx.x * blockDim.x + threadIdx.x;
  if (i < n) d[i] = (bf16)s[i];
}

__global__ void transpose_cast(const float* __restrict__ s, bf16* __restrict__ d,
                               int R, int C) {
  long i = (long)blockIdx.x * blockDim.x + threadIdx.x;
  if (i >= (long)R * C) return;
  int r = (int)(i / C), c = (int)(i % C);
  d[(long)c * R + r] = (bf16)s[i];
}

__global__ void transpose_W(const float* __restrict__ Wsrc, bf16* __restrict__ WT) {
  long i = (long)blockIdx.x * blockDim.x + threadIdx.x;
  if (i >= (long)K_ * N_ * N_) return;
  int k = (int)(i / ((long)N_ * N_));
  long rem = i - (long)k * N_ * N_;
  int j = (int)(rem / N_), n = (int)(rem % N_);
  WT[(long)k * N_ * N_ + (long)n * N_ + j] = (bf16)Wsrc[i];
}

// ---------- generic WMMA GEMM: C[M,Nout] = A[M,Kin] @ B  (B stored as [Nout][Kin]) ----------
// B panel (64 x TDM_CH) is DMA'd to LDS by the Tensor Data Mover, double-buffered:
// wave 0 issues chunk c+1, waits TENSORcnt<=1 for chunk c, all waves barrier + compute.
enum { MODE_FWD = 0, MODE_PLAIN = 1, MODE_DGRAD = 2, MODE_ACCSUB = 3 };

template<int MODE>
__launch_bounds__(128)
__global__ void gemm_k(const bf16* __restrict__ A, int lda,
                       const bf16* __restrict__ Bc, int ldb, long bHead,
                       int Kin,
                       const float* __restrict__ bias,
                       bf16* __restrict__ pre, int ldp,
                       bf16* __restrict__ out, int ldo, long oHead,
                       const bf16* __restrict__ dsrc, int ldd,
                       float* __restrict__ accp, int ldacc) {
  __shared__ bf16 Bsh[2][64 * TDM_ROWP];

  const int lane = threadIdx.x & 31;
  const int wave = threadIdx.x >> 5;
  const int m0 = blockIdx.y * 64 + wave * 16;
  const int n0 = blockIdx.x * 64;
  const int r15 = lane & 15, hi = lane >> 4, koff = hi * 8;

  Bc += (long)blockIdx.z * bHead;
  if (out) out += (long)blockIdx.z * oHead;

  v8f acc[4];
  #pragma unroll
  for (int f = 0; f < 4; ++f)
    #pragma unroll
    for (int e = 0; e < 8; ++e) acc[f][e] = 0.f;

  const bf16* ap = A + (long)(m0 + r15) * lda + koff;
  const bf16* btile = Bc + (long)n0 * ldb;           // 64-row panel start
  const unsigned lds0 = (unsigned)(unsigned long long)&Bsh[0][0];
  const unsigned lds1 = (unsigned)(unsigned long long)&Bsh[1][0];
  const int nc = Kin / TDM_CH;                       // 2 (Kin=256) or 4 (Kin=512)

  if (wave == 0) tdm_load_tile(btile, lds0, ldb);
  for (int c = 0; c < nc; ++c) {
    if (wave == 0) {
      if (c + 1 < nc) {
        tdm_load_tile(btile + (c + 1) * TDM_CH, ((c + 1) & 1) ? lds1 : lds0, ldb);
        __builtin_amdgcn_s_wait_tensorcnt(1);        // chunk c landed, c+1 in flight
      } else {
        __builtin_amdgcn_s_wait_tensorcnt(0);
      }
    }
    __syncthreads();                                 // chunk c visible to all waves
    const bf16* bl = &Bsh[c & 1][0];
    const bf16* apc = ap + c * TDM_CH;
    for (int kb = 0; kb < TDM_CH; kb += 32) {
      __builtin_prefetch(apc + kb + 128, 0, 1);
      v16bf a = frag_load(apc + kb);
      #pragma unroll
      for (int f = 0; f < 4; ++f) {
        v16bf b = frag_load(bl + (f * 16 + r15) * TDM_ROWP + koff + kb);  // ds_load_b128 x2
        acc[f] = __builtin_amdgcn_wmma_f32_16x16x32_bf16(
            false, a, false, b, (short)0, acc[f], false, false);
      }
    }
    __syncthreads();                                 // buffer reusable for DMA c+2
  }

  #pragma unroll
  for (int f = 0; f < 4; ++f) {
    #pragma unroll
    for (int r = 0; r < 8; ++r) {
      int m = m0 + r + 8 * hi;          // C layout: VGPR r holds M=r (+8 for hi lanes)
      int n = n0 + f * 16 + r15;        // lane&15 = N within fragment
      float c = acc[f][r];
      if (MODE == MODE_FWD) {
        c += bias[n];
        pre[(long)m * ldp + n] = (bf16)c;
        out[(long)m * ldo + n] = (bf16)(c * sigf(c));       // SiLU
      } else if (MODE == MODE_PLAIN) {
        out[(long)m * ldo + n] = (bf16)c;
      } else if (MODE == MODE_DGRAD) {
        float p = (float)dsrc[(long)m * ldd + n];
        out[(long)m * ldo + n] = (bf16)(c * dsilu(p));      // in-place safe: own element
      } else {                                              // MODE_ACCSUB
        accp[(long)m * ldacc + n] -= c;
      }
    }
  }
}

// ---------- fused x0-side bilinear gradient: out[b,n] = -sum_k alpha[b,k] * (W_k^T x1_b)[n] ----------
// Direct-global B path (exercises the clustered global_load_b128 + wmma pipeline).
__launch_bounds__(128)
__global__ void gemm_u_fused(const bf16* __restrict__ X1, const bf16* __restrict__ WT,
                             const float* __restrict__ alp, float* __restrict__ out) {
  const int lane = threadIdx.x & 31;
  const int wave = threadIdx.x >> 5;
  const int m0 = blockIdx.y * 64 + wave * 16;
  const int n0 = blockIdx.x * 64;
  const int r15 = lane & 15, hi = lane >> 4, koff = hi * 8;
  const int rowA = m0 + r15;

  v8f acc[4];
  #pragma unroll
  for (int f = 0; f < 4; ++f)
    #pragma unroll
    for (int e = 0; e < 8; ++e) acc[f][e] = 0.f;

  const bf16* ap = X1 + (long)rowA * N_ + koff;
  for (int k = 0; k < K_; ++k) {
    const float asc = alp[(long)rowA * K_ + k];   // per-row gate weight folded into A
    const bf16* wb = WT + (long)k * N_ * N_;
    for (int kb = 0; kb < N_; kb += 32) {
      v16bf a = frag_load(ap + kb);
      #pragma unroll
      for (int t = 0; t < 16; ++t) a[t] = (bf16)((float)a[t] * asc);
      #pragma unroll
      for (int f = 0; f < 4; ++f) {
        v16bf b = frag_load(wb + (long)(n0 + f * 16 + r15) * N_ + koff + kb);
        acc[f] = __builtin_amdgcn_wmma_f32_16x16x32_bf16(
            false, a, false, b, (short)0, acc[f], false, false);
      }
    }
  }

  #pragma unroll
  for (int f = 0; f < 4; ++f)
    #pragma unroll
    for (int r = 0; r < 8; ++r) {
      int m = m0 + r + 8 * hi, n = n0 + f * 16 + r15;
      out[(long)m * (2 * N_) + n] = -acc[f][r];
    }
}

// ---------- q[b,k] = dot(x1_b, V[b,k,:]) ; one wave32 per (b,k) ----------
__global__ void qreduce(const bf16* __restrict__ x1, const bf16* __restrict__ V,
                        float* __restrict__ q) {
  int lane = threadIdx.x & 31, wid = threadIdx.x >> 5;
  long g = (long)blockIdx.x * (blockDim.x >> 5) + wid;   // g = b*K + k
  if (g >= (long)B_ * K_) return;
  long b = g >> 3;
  const bf16* xp = x1 + b * N_;
  const bf16* vp = V + g * N_;
  float s = 0.f;
  for (int n = lane; n < N_; n += 32) s += (float)xp[n] * (float)vp[n];
  #pragma unroll
  for (int off = 16; off > 0; off >>= 1) s += __shfl_xor(s, off, 32);
  if (lane == 0) q[g] = s;
}

// ---------- gate: t = h2@w3+b3, alpha=softmax(t), galpha, gpre2_gt (in-place) ; one wave per row ----------
__global__ void gate_k(const bf16* __restrict__ h2, bf16* __restrict__ pre2,
                       const float* __restrict__ w3, const float* __restrict__ b3,
                       const float* __restrict__ q, float* __restrict__ alpha) {
  int lane = threadIdx.x & 31, wid = threadIdx.x >> 5;
  long b = (long)blockIdx.x * (blockDim.x >> 5) + wid;
  if (b >= B_) return;
  float t[8];
  #pragma unroll
  for (int j = 0; j < 8; ++j) t[j] = 0.f;
  const bf16* hp = h2 + b * H_;
  for (int c = lane; c < H_; c += 32) {
    float h = (float)hp[c];
    #pragma unroll
    for (int j = 0; j < 8; ++j) t[j] += h * w3[c * 8 + j];
  }
  #pragma unroll
  for (int off = 16; off > 0; off >>= 1)
    #pragma unroll
    for (int j = 0; j < 8; ++j) t[j] += __shfl_xor(t[j], off, 32);

  float mx = t[0] + b3[0];
  #pragma unroll
  for (int j = 0; j < 8; ++j) { t[j] += b3[j]; mx = fmaxf(mx, t[j]); }
  float e[8], s = 0.f;
  #pragma unroll
  for (int j = 0; j < 8; ++j) { e[j] = __expf(t[j] - mx); s += e[j]; }
  float inv = 1.f / s, al[8], qd = 0.f;
  #pragma unroll
  for (int j = 0; j < 8; ++j) { al[j] = e[j] * inv; qd += al[j] * q[b * K_ + j]; }
  float ga[8];
  #pragma unroll
  for (int j = 0; j < 8; ++j) ga[j] = al[j] * (q[b * K_ + j] - qd);
  if (lane == 0)
    #pragma unroll
    for (int j = 0; j < 8; ++j) alpha[b * K_ + j] = al[j];

  for (int c = lane; c < H_; c += 32) {          // g_pre2_gt, in-place (own element)
    float gh = 0.f;
    #pragma unroll
    for (int j = 0; j < 8; ++j) gh += ga[j] * w3[c * 8 + j];
    float p = (float)pre2[b * H_ + c];
    pre2[b * H_ + c] = (bf16)(gh * dsilu(p));
  }
}

// ---------- g_pre2 for g0/g1: w3[c]*silu'(pre2), in-place ----------
__global__ void gpre2_elem(bf16* __restrict__ pre2, const float* __restrict__ w3v, long n) {
  long i = (long)blockIdx.x * blockDim.x + threadIdx.x;
  if (i >= n) return;
  int c = (int)(i & (H_ - 1));
  float p = (float)pre2[i];
  pre2[i] = (bf16)(w3v[c] * dsilu(p));
}

// ---------- x1-side bilinear gradient: out[b, N+n] = -sum_k alpha[b,k]*V[b,k,n] ----------
__global__ void wsum_k(const float* __restrict__ alp, const bf16* __restrict__ V,
                       float* __restrict__ out) {
  long i = (long)blockIdx.x * blockDim.x + threadIdx.x;
  if (i >= (long)B_ * N_) return;
  long b = i >> 9; int n = (int)(i & (N_ - 1));
  float s = 0.f;
  #pragma unroll
  for (int k = 0; k < K_; ++k) s += alp[b * K_ + k] * (float)V[(b * K_ + k) * (long)N_ + n];
  out[b * (2L * N_) + N_ + n] = -s;
}

// ---------- host ----------
extern "C" void kernel_launch(void* const* d_in, const int* in_sizes, int n_in,
                              void* d_out_v, int out_size, void* d_ws, size_t ws_size,
                              hipStream_t stream) {
  (void)in_sizes; (void)n_in; (void)out_size; (void)ws_size;
  const float* z     = (const float*)d_in[0];
  const float* w1[3] = {(const float*)d_in[1],  (const float*)d_in[7],  (const float*)d_in[13]};
  const float* b1[3] = {(const float*)d_in[2],  (const float*)d_in[8],  (const float*)d_in[14]};
  const float* w2[3] = {(const float*)d_in[3],  (const float*)d_in[9],  (const float*)d_in[15]};
  const float* b2[3] = {(const float*)d_in[4],  (const float*)d_in[10], (const float*)d_in[16]};
  const float* w3[3] = {(const float*)d_in[5],  (const float*)d_in[11], (const float*)d_in[17]};
  const float* b3t   = (const float*)d_in[18];
  const float* W     = (const float*)d_in[19];
  float* d_out = (float*)d_out_v;

  char* base = (char*)d_ws; size_t off = 0;
  auto alloc = [&](size_t bytes) -> char* {
    char* p = base + off; off = (off + bytes + 255) & ~(size_t)255; return p;
  };
  bf16* x0b = (bf16*)alloc((size_t)B_ * N_ * 2);
  bf16* x1b = (bf16*)alloc((size_t)B_ * N_ * 2);
  bf16* Wb  = (bf16*)alloc((size_t)K_ * N_ * N_ * 2);
  bf16* WbT = (bf16*)alloc((size_t)K_ * N_ * N_ * 2);
  bf16 *w1T[3], *w2T[3], *w1c[3], *w2c[3], *pre1[3], *act1[3], *pre2[3];
  for (int i = 0; i < 3; ++i) {
    w1T[i] = (bf16*)alloc((size_t)N_ * H_ * 2);
    w2T[i] = (bf16*)alloc((size_t)H_ * H_ * 2);
    w1c[i] = (bf16*)alloc((size_t)N_ * H_ * 2);
    w2c[i] = (bf16*)alloc((size_t)H_ * H_ * 2);
  }
  for (int i = 0; i < 3; ++i) {
    pre1[i] = (bf16*)alloc((size_t)B_ * H_ * 2);
    act1[i] = (bf16*)alloc((size_t)B_ * H_ * 2);
    pre2[i] = (bf16*)alloc((size_t)B_ * H_ * 2);
  }
  bf16*  act2s = (bf16*)alloc((size_t)B_ * H_ * 2);   // h2 scratch (g0/g1, unused later)
  bf16*  act2t = (bf16*)alloc((size_t)B_ * H_ * 2);   // h2 for gt
  bf16*  Vb    = (bf16*)alloc((size_t)B_ * K_ * N_ * 2);
  float* qb    = (float*)alloc((size_t)B_ * K_ * 4);
  float* alp   = (float*)alloc((size_t)B_ * K_ * 4);

  const dim3 blk(128);
  const int T = 256;

  // ---- prep: casts / transposes ----
  split_cast_z<<<(B_ * N_ + T - 1) / T, T, 0, stream>>>(z, x0b, x1b);
  cast_flat<<<((long)K_ * N_ * N_ + T - 1) / T, T, 0, stream>>>(W, Wb, (long)K_ * N_ * N_);
  transpose_W<<<((long)K_ * N_ * N_ + T - 1) / T, T, 0, stream>>>(W, WbT);
  for (int i = 0; i < 3; ++i) {
    transpose_cast<<<(N_ * H_ + T - 1) / T, T, 0, stream>>>(w1[i], w1T[i], N_, H_);
    transpose_cast<<<(H_ * H_ + T - 1) / T, T, 0, stream>>>(w2[i], w2T[i], H_, H_);
    cast_flat<<<(N_ * H_ + T - 1) / T, T, 0, stream>>>(w1[i], w1c[i], (long)N_ * H_);
    cast_flat<<<(H_ * H_ + T - 1) / T, T, 0, stream>>>(w2[i], w2c[i], (long)H_ * H_);
  }

  // ---- MLP forwards (layer 3 of g0/g1 never needed for the gradient) ----
  const bf16* xin[3] = {x0b, x1b, x0b};
  bf16* h2p[3] = {act2s, act2s, act2t};
  for (int i = 0; i < 3; ++i)
    gemm_k<MODE_FWD><<<dim3(H_ / 64, B_ / 64, 1), blk, 0, stream>>>(
        xin[i], N_, w1T[i], N_, 0, N_, b1[i],
        pre1[i], H_, act1[i], H_, 0, nullptr, 0, nullptr, 0);
  for (int i = 0; i < 3; ++i)
    gemm_k<MODE_FWD><<<dim3(H_ / 64, B_ / 64, 1), blk, 0, stream>>>(
        act1[i], H_, w2T[i], H_, 0, H_, b2[i],
        pre2[i], H_, h2p[i], H_, 0, nullptr, 0, nullptr, 0);

  // ---- V[b,k,:] = W_k x0_b  (K batched GEMMs; B operand = W row-major directly) ----
  gemm_k<MODE_PLAIN><<<dim3(N_ / 64, B_ / 64, K_), blk, 0, stream>>>(
      x0b, N_, Wb, N_, (long)N_ * N_, N_, nullptr,
      nullptr, 0, Vb, K_ * N_, (long)N_, nullptr, 0, nullptr, 0);

  // ---- q, gate (alpha + gpre2_gt in-place) ----
  qreduce<<<(B_ * K_) / 8, 256, 0, stream>>>(x1b, Vb, qb);
  gate_k<<<B_ / 8, 256, 0, stream>>>(act2t, pre2[2], w3[2], b3t, qb, alp);

  // ---- gpre2 for g0/g1 (g_h2 = w3 broadcast), in-place ----
  gpre2_elem<<<((long)B_ * H_ + T - 1) / T, T, 0, stream>>>(pre2[0], w3[0], (long)B_ * H_);
  gpre2_elem<<<((long)B_ * H_ + T - 1) / T, T, 0, stream>>>(pre2[1], w3[1], (long)B_ * H_);

  // ---- bilinear gradients into d_out (x0 half fused over k; x1 half via weighted sum of V) ----
  gemm_u_fused<<<dim3(N_ / 64, B_ / 64, 1), blk, 0, stream>>>(x1b, WbT, alp, d_out);
  wsum_k<<<((long)B_ * N_ + T - 1) / T, T, 0, stream>>>(alp, Vb, d_out);

  // ---- dgrad: gpre1 = (gpre2 @ W2^T) * silu'(pre1), stored in-place into pre1 ----
  for (int i = 0; i < 3; ++i)
    gemm_k<MODE_DGRAD><<<dim3(H_ / 64, B_ / 64, 1), blk, 0, stream>>>(
        pre2[i], H_, w2c[i], H_, 0, H_, nullptr,
        nullptr, 0, pre1[i], H_, 0, pre1[i], H_, nullptr, 0);

  // ---- final: d_out -= gpre1 @ W1^T  (g0, gt -> x0 half; g1 -> x1 half) ----
  gemm_k<MODE_ACCSUB><<<dim3(N_ / 64, B_ / 64, 1), blk, 0, stream>>>(
      pre1[0], H_, w1c[0], H_, 0, H_, nullptr,
      nullptr, 0, nullptr, 0, 0, nullptr, 0, d_out, 2 * N_);
  gemm_k<MODE_ACCSUB><<<dim3(N_ / 64, B_ / 64, 1), blk, 0, stream>>>(
      pre1[2], H_, w1c[2], H_, 0, H_, nullptr,
      nullptr, 0, nullptr, 0, 0, nullptr, 0, d_out, 2 * N_);
  gemm_k<MODE_ACCSUB><<<dim3(N_ / 64, B_ / 64, 1), blk, 0, stream>>>(
      pre1[1], H_, w1c[1], H_, 0, H_, nullptr,
      nullptr, 0, nullptr, 0, 0, nullptr, 0, d_out + N_, 2 * N_);
}
